// FLSVIR_25975962206450
// MI455X (gfx1250) — matrix operand order repaired and checked by reading, not compiled, not run on hardware
//
#include <hip/hip_runtime.h>
#include <hip/hip_bf16.h>

// ---------------- problem constants ----------------
#define BB   32
#define NN   197
#define CC   768
#define KW   197
#define LL   12
#define HID  3072
#define HH   12
#define DD   64
#define BN   (BB*NN)          // 6304
#define C3   (3*CC)           // 2304

// ---------------- WMMA types ----------------
typedef __attribute__((ext_vector_type(16))) __bf16 v16bf;
typedef __attribute__((ext_vector_type(8)))  float  v8f;

union Frag {
    v16bf v;
    uint4 q[2];
};

// ---------------- helpers ----------------
__device__ __forceinline__ unsigned short f2bf(float f) {
    unsigned int u = __float_as_uint(f);
    unsigned int r = (u + 0x7FFFu + ((u >> 16) & 1u)) >> 16;
    return (unsigned short)r;
}

__device__ __forceinline__ float blockReduceSum(float v, float* red) {
    int t = threadIdx.x;
    red[t] = v; __syncthreads();
    for (int o = 128; o > 0; o >>= 1) {
        if (t < o) red[t] += red[t + o];
        __syncthreads();
    }
    float r = red[0]; __syncthreads();
    return r;
}

__device__ __forceinline__ float blockReduceMax(float v, float* red) {
    int t = threadIdx.x;
    red[t] = v; __syncthreads();
    for (int o = 128; o > 0; o >>= 1) {
        if (t < o) red[t] = fmaxf(red[t], red[t + o]);
        __syncthreads();
    }
    float r = red[0]; __syncthreads();
    return r;
}

// ---------------- elementwise kernels ----------------
__global__ void k_f2bf(const float* __restrict__ in, unsigned short* __restrict__ out, long long n) {
    long long i = (long long)blockIdx.x * blockDim.x + threadIdx.x;
    long long s = (long long)gridDim.x * blockDim.x;
    for (; i < n; i += s) out[i] = f2bf(in[i]);
}

// transpose-convert: in [R][Ccols] f32 -> out [Ccols][R] bf16
__global__ void k_f2bf_T(const float* __restrict__ in, unsigned short* __restrict__ out,
                         int R, int Ccols) {
    long long n = (long long)R * Ccols;
    long long i = (long long)blockIdx.x * blockDim.x + threadIdx.x;
    long long s = (long long)gridDim.x * blockDim.x;
    for (; i < n; i += s) {
        int r = (int)(i / Ccols), c = (int)(i % Ccols);
        out[(long long)c * R + r] = f2bf(in[i]);
    }
}

__global__ void k_gelu_bf(const float* __restrict__ in, unsigned short* __restrict__ out, long long n) {
    long long i = (long long)blockIdx.x * blockDim.x + threadIdx.x;
    long long s = (long long)gridDim.x * blockDim.x;
    for (; i < n; i += s) {
        float x = in[i];
        float g = 0.5f * x * (1.0f + erff(x * 0.70710678118654752f));
        out[i] = f2bf(g);
    }
}

__global__ void k_add_inplace(float* __restrict__ a, const float* __restrict__ b, long long n) {
    long long i = (long long)blockIdx.x * blockDim.x + threadIdx.x;
    long long s = (long long)gridDim.x * blockDim.x;
    for (; i < n; i += s) a[i] += b[i];
}

__global__ void k_avg(const float* __restrict__ a, const float* __restrict__ b,
                      float* __restrict__ o, long long n) {
    long long i = (long long)blockIdx.x * blockDim.x + threadIdx.x;
    long long s = (long long)gridDim.x * blockDim.x;
    for (; i < n; i += s) o[i] = 0.5f * (a[i] + b[i]);
}

// LayerNorm over last dim (len = CC), output bf16.  One block (256 thr) per row.
__global__ void k_ln_bf16(const float* __restrict__ x, const float* __restrict__ g,
                          const float* __restrict__ b, unsigned short* __restrict__ y, int len) {
    __shared__ float red[256];
    const float* xr = x + (long long)blockIdx.x * len;
    float v[3];
    float s = 0.f, sq = 0.f;
    #pragma unroll
    for (int i = 0; i < 3; i++) {
        int c = threadIdx.x + i * 256;
        v[i] = (c < len) ? xr[c] : 0.f;
        s += v[i]; sq += v[i] * v[i];
    }
    s  = blockReduceSum(s, red);
    sq = blockReduceSum(sq, red);
    float mu = s / (float)len;
    float var = sq / (float)len - mu * mu;
    float rs = rsqrtf(var + 1e-5f);
    unsigned short* yr = y + (long long)blockIdx.x * len;
    #pragma unroll
    for (int i = 0; i < 3; i++) {
        int c = threadIdx.x + i * 256;
        if (c < len) yr[c] = f2bf((v[i] - mu) * rs * g[c] + b[c]);
    }
}

// fused LayerNorm + softmax over last dim (len = KW <= 256), output bf16 probs.
__global__ void k_lnsoftmax_bf16(const float* __restrict__ x, const float* __restrict__ g,
                                 const float* __restrict__ b, unsigned short* __restrict__ y, int len) {
    __shared__ float red[256];
    const float* xr = x + (long long)blockIdx.x * len;
    int c = threadIdx.x;
    bool act = (c < len);
    float v = act ? xr[c] : 0.f;
    float s  = blockReduceSum(v, red);
    float sq = blockReduceSum(v * v, red);
    float mu = s / (float)len;
    float var = sq / (float)len - mu * mu;
    float rs = rsqrtf(var + 1e-5f);
    float ln = act ? ((v - mu) * rs * g[c] + b[c]) : -INFINITY;
    float mx = blockReduceMax(ln, red);
    float e = act ? __expf(ln - mx) : 0.f;
    float denom = blockReduceSum(e, red);
    if (act) y[(long long)blockIdx.x * len + c] = f2bf(e / denom);
}

// ---------------- batched WMMA GEMM ----------------
// C[z][m][n] = alpha * sum_k A'[m][k] * B'[k][n] (+ bias[n] or bias[m])
// batch z -> (zb, zh) = (z / Hdiv, z % Hdiv); per-matrix offsets zb*s?b + zh*s?h
// All per-batch matrices are < 2^31 elements: in-batch indexing is 32-bit.
struct GemmP {
    const unsigned short* A;
    const unsigned short* B;
    float* C;
    const float* bias;
    int M, N, K;
    int lda, ldb, ldc;
    long long sAb, sAh, sBb, sBh, sCb, sCh;
    int Hdiv;
    int biasMode;   // 0=none 1=per-col 2=per-row
    float alpha;
};

#define TM 64
#define TK 32
#define LSTR 40   // LDS row stride (bf16 elems); 80B keeps 16B alignment of 8-groups

// ---- fast GEMM: A row-major (lda along k), B pre-transposed (ldb along k) ----
// Interior blocks: branch-free double-buffered GLOBAL_LOAD_ASYNC_TO_LDS_B128 pipeline.
// Edge blocks: guarded scalar staging (32-bit addressing, per-row guards).
template<int TNF>
__launch_bounds__(128)
__global__ void k_gemm_async(GemmP p) {
    constexpr int NB = TNF / 32;     // 16-wide B subtiles per wave
    __shared__ unsigned short As[2][TM  * LSTR];
    __shared__ unsigned short Bs[2][TNF * LSTR];

    int z  = blockIdx.z;
    int zb = z / p.Hdiv, zh = z % p.Hdiv;
    const unsigned short* A  = p.A + zb * p.sAb + zh * p.sAh;
    const unsigned short* Bm = p.B + zb * p.sBb + zh * p.sBh;
    float* C = p.C + zb * p.sCb + zh * p.sCh;

    int m0 = blockIdx.y * TM, n0 = blockIdx.x * TNF;
    int tid = threadIdx.x;
    int lane = tid & 31, wid = tid >> 5;
    int wr = wid >> 1, wc = wid & 1;          // 2x2 waves; wave tile 32 x (TNF/2)
    int half = lane >> 4, l16 = lane & 15;

    v8f acc[2][NB] = {};

    auto compute = [&](int buf) {
        Frag fa[2], fb[NB];
        #pragma unroll
        for (int a = 0; a < 2; a++) {
            int r = wr * 32 + a * 16 + l16;
            fa[a].q[0] = *(const uint4*)&As[buf][r * LSTR + 8 * half];
            fa[a].q[1] = *(const uint4*)&As[buf][r * LSTR + 16 + 8 * half];
        }
        #pragma unroll
        for (int b = 0; b < NB; b++) {
            int cidx = wc * (TNF / 2) + b * 16 + l16;
            fb[b].q[0] = *(const uint4*)&Bs[buf][cidx * LSTR + 8 * half];
            fb[b].q[1] = *(const uint4*)&Bs[buf][cidx * LSTR + 16 + 8 * half];
        }
        #pragma unroll
        for (int a = 0; a < 2; a++)
            #pragma unroll
            for (int b = 0; b < NB; b++)
                acc[a][b] = __builtin_amdgcn_wmma_f32_16x16x32_bf16(
                    false, fa[a].v, false, fb[b].v, (short)0, acc[a][b], false, false);
    };

    bool interior = (m0 + TM <= p.M) && (n0 + TNF <= p.N);

    if (interior) {
        auto stageAsync = [&](int k0, int buf) {
            #pragma unroll
            for (int c = tid; c < TM * 4; c += 128) {         // 16B chunks of A
                int r = c >> 2, seg = (c & 3) << 3;
                unsigned voff = (unsigned)((m0 + r) * p.lda + k0 + seg) * 2u;
                unsigned laddr = (unsigned)(unsigned long long)&As[buf][r * LSTR + seg];
                asm volatile("global_load_async_to_lds_b128 %0, %1, %2"
                             :: "v"(laddr), "v"(voff), "s"(A) : "memory");
            }
            #pragma unroll
            for (int c = tid; c < TNF * 4; c += 128) {        // 16B chunks of B
                int r = c >> 2, seg = (c & 3) << 3;
                unsigned voff = (unsigned)((n0 + r) * p.ldb + k0 + seg) * 2u;
                unsigned laddr = (unsigned)(unsigned long long)&Bs[buf][r * LSTR + seg];
                asm volatile("global_load_async_to_lds_b128 %0, %1, %2"
                             :: "v"(laddr), "v"(voff), "s"(Bm) : "memory");
            }
        };
        auto stageTail = [&](int k0, int buf) {               // partial K tile, zero-filled
            for (int base = tid * 8; base < TM * TK; base += 128 * 8) {
                int r = base / TK, kk = base % TK;
                const unsigned short* Ar = A + (unsigned)((m0 + r) * p.lda + k0 + kk);
                int kRem = p.K - k0 - kk;                     // valid elements in this 8-group
                #pragma unroll
                for (int j = 0; j < 8; j++)
                    As[buf][r * LSTR + kk + j] = (j < kRem) ? Ar[j] : (unsigned short)0;
            }
            for (int base = tid * 8; base < TNF * TK; base += 128 * 8) {
                int r = base / TK, kk = base % TK;
                const unsigned short* Br = Bm + (unsigned)((n0 + r) * p.ldb + k0 + kk);
                int kRem = p.K - k0 - kk;
                #pragma unroll
                for (int j = 0; j < 8; j++)
                    Bs[buf][r * LSTR + kk + j] = (j < kRem) ? Br[j] : (unsigned short)0;
            }
        };

        int kFull = p.K & ~(TK - 1);
        int buf = 0;
        stageAsync(0, 0);
        asm volatile("s_wait_asynccnt 0x0" ::: "memory");
        __syncthreads();
        for (int k0 = 0; k0 < kFull; k0 += TK) {
            int nk = k0 + TK;
            if (nk < kFull)           stageAsync(nk, buf ^ 1);
            else if (nk < p.K)        stageTail(nk, buf ^ 1);
            compute(buf);
            asm volatile("s_wait_asynccnt 0x0" ::: "memory");
            __syncthreads();
            buf ^= 1;
        }
        if (kFull < p.K) compute(buf);
    } else {
        for (int k0 = 0; k0 < p.K; k0 += TK) {
            __syncthreads();
            for (int base = tid * 8; base < TM * TK; base += 128 * 8) {
                int r = base / TK, kk = base % TK;
                int gm = m0 + r;
                bool mOk = gm < p.M;
                const unsigned short* Ar = A + (unsigned)(gm * p.lda + k0 + kk);
                int kRem = p.K - k0 - kk;
                #pragma unroll
                for (int j = 0; j < 8; j++)
                    As[0][r * LSTR + kk + j] = (mOk && j < kRem) ? Ar[j] : (unsigned short)0;
            }
            for (int base = tid * 8; base < TNF * TK; base += 128 * 8) {
                int r = base / TK, kk = base % TK;
                int gn = n0 + r;
                bool nOk = gn < p.N;
                const unsigned short* Br = Bm + (unsigned)(gn * p.ldb + k0 + kk);
                int kRem = p.K - k0 - kk;
                #pragma unroll
                for (int j = 0; j < 8; j++)
                    Bs[0][r * LSTR + kk + j] = (nOk && j < kRem) ? Br[j] : (unsigned short)0;
            }
            __syncthreads();
            compute(0);
        }
    }

    #pragma unroll
    for (int a = 0; a < 2; a++) {
        #pragma unroll
        for (int b = 0; b < NB; b++) {
            int n = n0 + wc * (TNF / 2) + b * 16 + l16;
            if (n >= p.N) continue;
            float bcol = (p.biasMode == 1) ? p.bias[n] : 0.f;
            #pragma unroll
            for (int i = 0; i < 8; i++) {
                int m = m0 + wr * 32 + a * 16 + 8 * half + i;
                if (m >= p.M) continue;
                float v = p.alpha * acc[a][b][i] + bcol;
                if (p.biasMode == 2) v += p.bias[m];
                C[(unsigned)(m * p.ldc + n)] = v;
            }
        }
    }
}

// ---- scalar GEMM: A accessed transposed (A'[m][k]=A[k*lda+m]), B row-major [k][n] ----
__launch_bounds__(128)
__global__ void k_gemm_ta(GemmP p) {
    __shared__ unsigned short As[TM * LSTR];
    __shared__ unsigned short Bs[64 * LSTR];

    int z  = blockIdx.z;
    int zb = z / p.Hdiv, zh = z % p.Hdiv;
    const unsigned short* A  = p.A + zb * p.sAb + zh * p.sAh;
    const unsigned short* Bm = p.B + zb * p.sBb + zh * p.sBh;
    float* C = p.C + zb * p.sCb + zh * p.sCh;

    int m0 = blockIdx.y * TM, n0 = blockIdx.x * 64;
    int tid = threadIdx.x;
    int lane = tid & 31, wid = tid >> 5;
    int wr = wid >> 1, wc = wid & 1;
    int half = lane >> 4, l16 = lane & 15;

    v8f acc[2][2] = {};

    for (int k0 = 0; k0 < p.K; k0 += TK) {
        __syncthreads();
        // A'[m][k] = A[k*lda + m]: strided along k (transpose-on-stage)
        for (int base = tid * 8; base < TM * TK; base += 128 * 8) {
            int r = base / TK, kk = base % TK;
            int gm = m0 + r;
            bool mOk = gm < p.M;
            const unsigned short* Ac = A + (unsigned)((k0 + kk) * p.lda + gm);
            int kRem = p.K - k0 - kk;
            #pragma unroll
            for (int j = 0; j < 8; j++)
                As[r * LSTR + kk + j] =
                    (mOk && j < kRem) ? Ac[(unsigned)(j * p.lda)] : (unsigned short)0;
        }
        for (int base = tid * 8; base < 64 * TK; base += 128 * 8) {
            int r = base / TK, kk = base % TK;
            int gn = n0 + r;
            bool nOk = gn < p.N;
            const unsigned short* Bc = Bm + (unsigned)((k0 + kk) * p.ldb + gn);
            int kRem = p.K - k0 - kk;
            #pragma unroll
            for (int j = 0; j < 8; j++)
                Bs[r * LSTR + kk + j] =
                    (nOk && j < kRem) ? Bc[(unsigned)(j * p.ldb)] : (unsigned short)0;
        }
        __syncthreads();

        Frag fa[2], fb[2];
        #pragma unroll
        for (int a = 0; a < 2; a++) {
            int r = wr * 32 + a * 16 + l16;
            fa[a].q[0] = *(const uint4*)&As[r * LSTR + 8 * half];
            fa[a].q[1] = *(const uint4*)&As[r * LSTR + 16 + 8 * half];
        }
        #pragma unroll
        for (int b = 0; b < 2; b++) {
            int cidx = wc * 32 + b * 16 + l16;
            fb[b].q[0] = *(const uint4*)&Bs[cidx * LSTR + 8 * half];
            fb[b].q[1] = *(const uint4*)&Bs[cidx * LSTR + 16 + 8 * half];
        }
        #pragma unroll
        for (int a = 0; a < 2; a++)
            #pragma unroll
            for (int b = 0; b < 2; b++)
                acc[a][b] = __builtin_amdgcn_wmma_f32_16x16x32_bf16(
                    false, fa[a].v, false, fb[b].v, (short)0, acc[a][b], false, false);
    }

    #pragma unroll
    for (int a = 0; a < 2; a++) {
        #pragma unroll
        for (int b = 0; b < 2; b++) {
            int n = n0 + wc * 32 + b * 16 + l16;
            if (n >= p.N) continue;
            float bcol = (p.biasMode == 1) ? p.bias[n] : 0.f;
            #pragma unroll
            for (int i = 0; i < 8; i++) {
                int m = m0 + wr * 32 + a * 16 + 8 * half + i;
                if (m >= p.M) continue;
                float v = p.alpha * acc[a][b][i] + bcol;
                if (p.biasMode == 2) v += p.bias[m];
                C[(unsigned)(m * p.ldc + n)] = v;
            }
        }
    }
}

// ---------------- host orchestration ----------------
static inline int ceil_div(int a, int b) { return (a + b - 1) / b; }

extern "C" void kernel_launch(void* const* d_in, const int* in_sizes, int n_in,
                              void* d_out, int out_size, void* d_ws, size_t ws_size,
                              hipStream_t stream) {
    const float* x      = (const float*)d_in[0];
    const float* qkv_w  = (const float*)d_in[1];
    const float* fc_w   = (const float*)d_in[2];
    const float* fc_b   = (const float*)d_in[3];
    const float* aln_g  = (const float*)d_in[4];
    const float* aln_b  = (const float*)d_in[5];
    const float* fc2_w  = (const float*)d_in[6];
    const float* fc2_b  = (const float*)d_in[7];
    const float* w1     = (const float*)d_in[8];
    const float* b1     = (const float*)d_in[9];
    const float* w2     = (const float*)d_in[10];
    const float* b2     = (const float*)d_in[11];
    const float* fln_g  = (const float*)d_in[12];
    const float* fln_b  = (const float*)d_in[13];
    const float* gln_g  = (const float*)d_in[14];
    const float* gln_b  = (const float*)d_in[15];
    (void)in_sizes; (void)n_in; (void)out_size; (void)ws_size;

    // ---- workspace bump allocator ----
    char* w = (char*)d_ws;
    size_t off = 0;
    auto alloc = [&](size_t bytes) -> void* {
        void* p = w + off;
        off += (bytes + 255) & ~(size_t)255;
        return p;
    };
    unsigned short* qkvw_bf = (unsigned short*)alloc((size_t)CC * C3 * 2);   // [C3][CC] (transposed)
    unsigned short* fcw_bf  = (unsigned short*)alloc((size_t)NN * KW * 2);   // [KW][NN] (transposed)
    unsigned short* fc2w_bf = (unsigned short*)alloc((size_t)KW * NN * 2);   // [KW][NN] (natural)
    unsigned short* w1_bf   = (unsigned short*)alloc((size_t)CC * HID * 2);  // [HID][CC]
    unsigned short* w2_bf   = (unsigned short*)alloc((size_t)HID * CC * 2);  // [CC][HID]
    float* x1       = (float*)alloc((size_t)BN * CC * 4);
    float* x2       = (float*)alloc((size_t)BN * CC * 4);
    unsigned short* ln_bf = (unsigned short*)alloc((size_t)BN * CC * 2);
    float* R2       = (float*)alloc((size_t)BN * C3 * 4);   // qkv_f32 | y_f32+attn_out | mlp_out
    unsigned short* qkv_bf = (unsigned short*)alloc((size_t)BN * C3 * 2);
    float* R1       = (float*)alloc((size_t)BN * HID * 4);  // scores/t | mlp hidden
    unsigned short* sc_bf = (unsigned short*)alloc((size_t)BB * HH * NN * NN * 2);
    unsigned short* p_bf  = (unsigned short*)alloc((size_t)BB * HH * NN * KW * 2);
    unsigned short* y_bf  = (unsigned short*)alloc((size_t)BB * HH * KW * DD * 2);
    unsigned short* hid_bf = (unsigned short*)alloc((size_t)BN * HID * 2);

    float* y_f32    = R2;                        // BN*CC floats (== B*H*KW*DD)
    float* attn_out = R2 + (size_t)BN * CC;      // BN*CC floats
    float* mlp_out  = R2;                        // reused after y_f32 dead

    const int EB = 2048;

    auto mkP = [](const unsigned short* A, const unsigned short* Bp, float* Cp,
                  const float* bias, int biasMode, int M, int Nn, int K,
                  int lda, int ldb, int ldc,
                  long long sAb, long long sAh, long long sBb, long long sBh,
                  long long sCb, long long sCh, int Hdiv, float alpha) {
        GemmP pp;
        pp.A = A; pp.B = Bp; pp.C = Cp; pp.bias = bias;
        pp.M = M; pp.N = Nn; pp.K = K;
        pp.lda = lda; pp.ldb = ldb; pp.ldc = ldc;
        pp.sAb = sAb; pp.sAh = sAh; pp.sBb = sBb; pp.sBh = sBh; pp.sCb = sCb; pp.sCh = sCh;
        pp.Hdiv = Hdiv; pp.biasMode = biasMode; pp.alpha = alpha;
        return pp;
    };
    auto gemmF128 = [&](GemmP pp, int batches) {   // wide-N fast path
        dim3 g(ceil_div(pp.N, 128), ceil_div(pp.M, TM), batches);
        k_gemm_async<128><<<g, dim3(128), 0, stream>>>(pp);
    };
    auto gemmF64 = [&](GemmP pp, int batches) {    // narrow-N fast path
        dim3 g(ceil_div(pp.N, 64), ceil_div(pp.M, TM), batches);
        k_gemm_async<64><<<g, dim3(128), 0, stream>>>(pp);
    };
    auto gemmT = [&](GemmP pp, int batches) {      // A transposed, B row-major (scalar)
        dim3 g(ceil_div(pp.N, 64), ceil_div(pp.M, TM), batches);
        k_gemm_ta<<<g, dim3(128), 0, stream>>>(pp);
    };

    // ---- one-time weight conversion (transposed to K-major) + x1=x2=x ----
    k_f2bf_T<<<EB, 256, 0, stream>>>(qkv_w, qkvw_bf, CC, C3);
    k_f2bf_T<<<EB, 256, 0, stream>>>(fc_w,  fcw_bf,  NN, KW);
    k_f2bf  <<<EB, 256, 0, stream>>>(fc2_w, fc2w_bf, (long long)KW * NN);
    k_f2bf_T<<<EB, 256, 0, stream>>>(w1,    w1_bf,   CC, HID);
    k_f2bf_T<<<EB, 256, 0, stream>>>(w2,    w2_bf,   HID, CC);
    hipMemcpyAsync(x1, x, (size_t)BN * CC * 4, hipMemcpyDeviceToDevice, stream);
    hipMemcpyAsync(x2, x, (size_t)BN * CC * 4, hipMemcpyDeviceToDevice, stream);

    const long long NN2 = (long long)NN * NN;
    for (int i = 0; i < LL; i++) {
        // ---- f: attention(LN(x2)) ----
        k_ln_bf16<<<BN, 256, 0, stream>>>(x2, fln_g + (size_t)i * CC, fln_b + (size_t)i * CC, ln_bf, CC);
        // qkv = ln @ qkv_w   [BN,C3]
        gemmF128(mkP(ln_bf, qkvw_bf, R2, nullptr, 0, BN, C3, CC, CC, CC, C3,
                     0,0, 0,0, 0,0, 1, 1.0f), 1);
        k_f2bf<<<EB, 256, 0, stream>>>(R2, qkv_bf, (long long)BN * C3);
        // scores = q @ k^T * d^-0.5    (384 batches, 197x197x64); k rows are K-major already
        gemmF64(mkP(qkv_bf, qkv_bf + CC, R1, nullptr, 0, NN, NN, DD,
                    C3, C3, NN,
                    (long long)NN * C3, DD, (long long)NN * C3, DD,
                    (long long)HH * NN2, NN2, HH, 0.125f), BB * HH);
        k_f2bf<<<EB, 256, 0, stream>>>(R1, sc_bf, (long long)BB * HH * NN2);
        // t = scores @ fc_w + fc_b     (384 batches, 197x197x197), fc_w pre-transposed
        gemmF64(mkP(sc_bf, fcw_bf, R1, fc_b, 1, NN, KW, NN,
                    NN, NN, KW,
                    NN2, 0, 0, 0, NN2, 0, 1, 1.0f), BB * HH);
        // p = softmax(LN(t)) over KW
        k_lnsoftmax_bf16<<<BB * HH * NN, 256, 0, stream>>>(R1, aln_g, aln_b, p_bf, KW);
        // y = p^T @ v                  (384 batches, 197x64x197)
        gemmT(mkP(p_bf, qkv_bf + 2 * CC, y_f32, nullptr, 0, KW, DD, NN,
                  KW, C3, DD,
                  (long long)HH * NN2, NN2, (long long)NN * C3, DD,
                  (long long)HH * KW * DD, (long long)KW * DD, HH, 1.0f), BB * HH);
        k_f2bf<<<EB, 256, 0, stream>>>(y_f32, y_bf, (long long)BB * HH * KW * DD);
        // attn_out[b,n,h,d] = fc2_w^T @ y + fc2_b[n]   (384 batches, 197x64x197)
        gemmT(mkP(fc2w_bf, y_bf, attn_out, fc2_b, 2, NN, DD, KW,
                  NN, DD, CC,
                  0, 0, (long long)HH * KW * DD, (long long)KW * DD,
                  (long long)NN * CC, DD, HH, 1.0f), BB * HH);
        // y1 = x1 + f(x2)
        k_add_inplace<<<EB, 256, 0, stream>>>(x1, attn_out, (long long)BN * CC);

        // ---- g: mlp(LN(y1)) ----
        k_ln_bf16<<<BN, 256, 0, stream>>>(x1, gln_g + (size_t)i * CC, gln_b + (size_t)i * CC, ln_bf, CC);
        gemmF128(mkP(ln_bf, w1_bf, R1, b1, 1, BN, HID, CC, CC, CC, HID,
                     0,0, 0,0, 0,0, 1, 1.0f), 1);
        k_gelu_bf<<<EB, 256, 0, stream>>>(R1, hid_bf, (long long)BN * HID);
        gemmF128(mkP(hid_bf, w2_bf, mlp_out, b2, 1, BN, CC, HID, HID, HID, CC,
                     0,0, 0,0, 0,0, 1, 1.0f), 1);
        // y2 = x2 + g(y1)
        k_add_inplace<<<EB, 256, 0, stream>>>(x2, mlp_out, (long long)BN * CC);
    }

    // out = (x1 + x2) * 0.5
    k_avg<<<EB, 256, 0, stream>>>(x1, x2, (float*)d_out, (long long)BN * CC);
}
